// MaxwellFFNNModel_36155034698237
// MI455X (gfx1250) — compile-verified, matrix operand order checked by reading, not run on hardware
//
#include <hip/hip_runtime.h>
#include <hip/hip_bf16.h>
#include <hip/hip_fp16.h>

typedef __attribute__((ext_vector_type(16))) _Float16 v16h;
typedef __attribute__((ext_vector_type(8)))  float    v8f;

#define HID   64
#define TLEN  1024
#define E_MOD      2.0f
#define E_INF      0.5f

// CDNA5 has a native transcendental v_tanh_f32 (single TRANS32 op).
// Use the clang builtin when declared; otherwise emit it via inline asm
// with a trailing v_nop to satisfy the TRANS-op result-use hazard rule.
__device__ __forceinline__ float fast_tanh(float x) {
#if __has_builtin(__builtin_amdgcn_tanhf)
    return __builtin_amdgcn_tanhf(x);
#elif __has_builtin(__builtin_amdgcn_tanh_f32)
    return __builtin_amdgcn_tanh_f32(x);
#else
    float r;
    asm volatile("v_tanh_f32 %0, %1\n\tv_nop" : "=v"(r) : "v"(x));
    return r;
#endif
}

// 16-bit A-matrix 16x32 layout (ISA 7.12.2): element e (0..15) of the per-lane
// v16h fragment holds K = kmap(e, laneHi):
//   lanes 0-15 : K in {0..7, 16..23};  lanes 16-31 : K in {8..15, 24..31}
__device__ __forceinline__ int kmap(int e, int laneHi) {
    return laneHi ? ((e < 8) ? (8 + e) : (16 + e))
                  : ((e < 8) ?  e      : ( 8 + e));
}

__global__ __launch_bounds__(256) void
maxwell_ffnn_scan_kernel(const float* __restrict__ x,   // (B, T, 2)
                         const float* __restrict__ W1,  // (2, 64)
                         const float* __restrict__ b1,  // (64)
                         const float* __restrict__ W2,  // (64, 64)
                         const float* __restrict__ b2,  // (64)
                         const float* __restrict__ W3,  // (64, 1)
                         const float* __restrict__ b3,  // (1)
                         float* __restrict__ out,       // (B, T, 1)
                         int B)
{
    const int lane   = threadIdx.x & 31;
    const int waveId = threadIdx.x >> 5;
    const int laneHi = lane >> 4;          // 0: lanes 0-15, 1: lanes 16-31
    const int m      = lane & 15;          // batch row within tile (A-matrix M)
    const int tile   = blockIdx.x * (blockDim.x >> 5) + waveId;
    const int row0   = tile * 16;
    if (row0 >= B) return;

    // ---- loop-invariant register-resident weights -------------------------
    // Layer-1 weights/bias, replicated into the exact A-fragment element order.
    float w1e[2][16], w1g[2][16], bb1[2][16];
#pragma unroll
    for (int f = 0; f < 2; ++f)
#pragma unroll
        for (int e = 0; e < 16; ++e) {
            int k = 32 * f + kmap(e, laneHi);
            w1e[f][e] = W1[k];        // row 0 of W1: multiplies eps
            w1g[f][e] = W1[HID + k];  // row 1 of W1: multiplies gamma
            bb1[f][e] = b1[k];
        }

    // W2 as 8 B-matrix fragments (kt = K-half 0..1, nt = N-tile 0..3),
    // f16, resident for the whole scan.
    v16h bfrag[2][4];
#pragma unroll
    for (int kt = 0; kt < 2; ++kt)
#pragma unroll
        for (int nt = 0; nt < 4; ++nt)
#pragma unroll
            for (int e = 0; e < 16; ++e) {
                int k = 32 * kt + kmap(e, laneHi);
                int n = 16 * nt + m;
                bfrag[kt][nt][e] = (_Float16)W2[k * HID + n];
            }

    // b2 / W3 in C/D layout: lane owns column n = 16*nt + m of each N-tile.
    float b2sel[4], w3sel[4];
#pragma unroll
    for (int nt = 0; nt < 4; ++nt) {
        b2sel[nt] = b2[16 * nt + m];
        w3sel[nt] = W3[16 * nt + m];
    }
    const float b3v = b3[0];

    // ---- scan state -------------------------------------------------------
    float gam = 0.0f;
    const size_t xbase = (size_t)(row0 + m) * TLEN * 2;  // this lane's row
    const size_t obase = (size_t)(row0 + m) * TLEN;

    for (int t = 0; t < TLEN; ++t) {
        // x_t = (eps, dt) for this lane's batch row
        const float2 xv = *reinterpret_cast<const float2*>(x + xbase + 2 * (size_t)t);
        const float eps = xv.x;
        const float dtn = xv.y;

        // ---- layer 1: h1 = tanh(eps*W1[0] + gam*W1[1] + b1), directly in
        // WMMA A-fragment layout (each lane computes exactly its 32 K values)
        v16h afrag[2];
#pragma unroll
        for (int f = 0; f < 2; ++f)
#pragma unroll
            for (int e = 0; e < 16; ++e) {
                float pre = fmaf(eps, w1e[f][e], fmaf(gam, w1g[f][e], bb1[f][e]));
                afrag[f][e] = (_Float16)fast_tanh(pre);
            }

        // ---- layer 2: acc = h1 @ W2 + b2  (8x v_wmma_f32_16x16x32_f16)
        v8f acc[4];
#pragma unroll
        for (int nt = 0; nt < 4; ++nt) {
#pragma unroll
            for (int r = 0; r < 8; ++r) acc[nt][r] = b2sel[nt];   // seed with bias
            acc[nt] = __builtin_amdgcn_wmma_f32_16x16x32_f16(
                false, afrag[0], false, bfrag[0][nt], (short)0, acc[nt], false, false);
            acc[nt] = __builtin_amdgcn_wmma_f32_16x16x32_f16(
                false, afrag[1], false, bfrag[1][nt], (short)0, acc[nt], false, false);
        }

        // ---- layer 3: per-lane partial of tanh(acc) . W3, per C/D row r
        // C/D layout: VGPR r, lanes 0-15 -> row r, lanes 16-31 -> row r+8
        float part[8];
#pragma unroll
        for (int r = 0; r < 8; ++r) {
            float p = 0.0f;
#pragma unroll
            for (int nt = 0; nt < 4; ++nt)
                p = fmaf(fast_tanh(acc[nt][r]), w3sel[nt], p);
            part[r] = p;
        }

        // butterfly-reduce within each 16-lane half (rows 0-7 in lanes 0-15,
        // rows 8-15 in lanes 16-31), then swap halves for the redistribution
        float partS[8];
#pragma unroll
        for (int r = 0; r < 8; ++r) {
            float p = part[r];
            p += __shfl_xor(p, 1);
            p += __shfl_xor(p, 2);
            p += __shfl_xor(p, 4);
            p += __shfl_xor(p, 8);
            part[r]  = p;
            partS[r] = __shfl_xor(p, 16);
        }

        // each lane picks gamma_dot for its own row m (unrolled cndmask select)
        const bool rowHi = (m >= 8);
        float gdot = 0.0f;
#pragma unroll
        for (int r = 0; r < 8; ++r) {
            float v = (rowHi == (bool)laneHi) ? part[r] : partS[r];
            gdot = ((m & 7) == r) ? v : gdot;
        }
        gdot += b3v;

        // ---- state update + output -------------------------------------
        gam = fmaf(dtn, gdot, gam);
        // sigma = E_INF*eps + E_MOD*(eps - gam) = 2.5*eps - 2*gam
        const float sig = fmaf(E_INF + E_MOD, eps, -E_MOD * gam);
        if (lane < 16) out[obase + (size_t)t] = sig;
    }
}

extern "C" void kernel_launch(void* const* d_in, const int* in_sizes, int n_in,
                              void* d_out, int out_size, void* d_ws, size_t ws_size,
                              hipStream_t stream) {
    const float* x  = (const float*)d_in[0];
    const float* W1 = (const float*)d_in[1];
    const float* b1 = (const float*)d_in[2];
    const float* W2 = (const float*)d_in[3];
    const float* b2 = (const float*)d_in[4];
    const float* W3 = (const float*)d_in[5];
    const float* b3 = (const float*)d_in[6];
    float* out = (float*)d_out;

    const int B = in_sizes[0] / (TLEN * 2);      // 2048
    const int wavesPerBlock = 8;                 // 256 threads = 8 wave32
    const int tiles = (B + 15) / 16;             // 128 independent wave-tiles
    const int grid = (tiles + wavesPerBlock - 1) / wavesPerBlock;

    maxwell_ffnn_scan_kernel<<<grid, wavesPerBlock * 32, 0, stream>>>(
        x, W1, b1, W2, b2, W3, b3, out, B);
}
